// KnowledgeAttention_42339787604842
// MI455X (gfx1250) — matrix operand order repaired and checked by reading, not compiled, not run on hardware
//
#include <hip/hip_runtime.h>
#include <cstdint>

// ---------------------------------------------------------------------------
// Types
// ---------------------------------------------------------------------------
typedef __bf16 bf16_t;
typedef __attribute__((ext_vector_type(16))) __bf16 v16bf;
typedef __attribute__((ext_vector_type(8)))  float  v8f;
typedef __attribute__((ext_vector_type(4)))  unsigned int u32x4;
typedef __attribute__((ext_vector_type(4)))  float  f32x4;
typedef __attribute__((ext_vector_type(4)))  unsigned short u16x4;

union FragB {
    v16bf v;
    unsigned short u[16];
    u32x4 q[2];
};

union Stage16 {            // 16 halfs (32B) of staged global data
    u32x4 q[2];
    unsigned short u[16];
};

__device__ __forceinline__ bf16_t f2bf(float f) {
    union { float f; unsigned u; } c; c.f = f;
    unsigned r = c.u + 0x7FFFu + ((c.u >> 16) & 1u);   // round-to-nearest-even
    union { unsigned short s; bf16_t b; } o;
    o.s = (unsigned short)(r >> 16);
    return o.b;
}

__device__ __forceinline__ v8f wmma_bf16(v16bf a, v16bf b, v8f c) {
    // D = A(16x32 bf16) * B(32x16 bf16) + C(16x16 f32)
    return __builtin_amdgcn_wmma_f32_16x16x32_bf16(false, a, false, b,
                                                   (short)0, c, false, false);
}

// gfx1250 async DMA: global memory -> LDS, 16B per lane, tracked by ASYNCcnt.
__device__ __forceinline__ void async_copy_b128(unsigned lds_off, const void* g) {
    asm volatile("global_load_async_to_lds_b128 %0, %1, off"
                 :: "v"(lds_off), "v"(g) : "memory");
}
__device__ __forceinline__ void wait_async0(void) {
    asm volatile("s_wait_asynccnt 0" ::: "memory");
}
__device__ __forceinline__ unsigned lds_off32(const void* p) {
    return (unsigned)(size_t)p;   // flat LDS address: low 32 bits = wg-relative offset
}

// ds_swizzle XOR within 32-lane group: offset[14:10]=xor, [4:0]=and=0x1f
template <int XORM>
__device__ __forceinline__ float ds_swz_f(float x) {
    union { float f; int i; } u; u.f = x;
    u.i = __builtin_amdgcn_ds_swizzle(u.i, 0x1F | (XORM << 10));
    return u.f;
}

// max / sum reduction across a 16-lane half-group (rows live per half-group)
__device__ __forceinline__ float red16_max(float t) {
    t = fmaxf(t, ds_swz_f<1>(t));
    t = fmaxf(t, ds_swz_f<2>(t));
    t = fmaxf(t, ds_swz_f<4>(t));
    t = fmaxf(t, ds_swz_f<8>(t));
    return t;
}
__device__ __forceinline__ float red16_sum(float t) {
    t += ds_swz_f<1>(t);
    t += ds_swz_f<2>(t);
    t += ds_swz_f<4>(t);
    t += ds_swz_f<8>(t);
    return t;
}

#define T_SEQ  2048
#define K_SEQ  512
#define DMODEL 768
#define NHEAD  12
#define HDIM   64

// ---------------------------------------------------------------------------
// f32 -> bf16 convert (vectorized x4)
// ---------------------------------------------------------------------------
__global__ __launch_bounds__(256)
void cvt_f32_bf16_kernel(const float* __restrict__ src, bf16_t* __restrict__ dst, int n4) {
    int i = blockIdx.x * 256 + threadIdx.x;
    if (i < n4) {
        f32x4 v = *(const f32x4*)(src + (size_t)i * 4);
        union { u16x4 p; bf16_t b[4]; } o;
        o.b[0] = f2bf(v.x); o.b[1] = f2bf(v.y); o.b[2] = f2bf(v.z); o.b[3] = f2bf(v.w);
        *(u16x4*)(dst + (size_t)i * 4) = o.p;
    }
}

// ---------------------------------------------------------------------------
// cosine similarity: one wave per (b,k)
// ---------------------------------------------------------------------------
__global__ __launch_bounds__(256)
void sim_kernel(const float* __restrict__ pooled, const float* __restrict__ kg_key,
                float* __restrict__ sim) {
    int gw   = (blockIdx.x * 256 + threadIdx.x) >> 5;   // 0..4095
    int lane = threadIdx.x & 31;
    int b = gw >> 9;
    int k = gw & (K_SEQ - 1);
    const float* p  = pooled + (size_t)b * DMODEL;
    const float* kk = kg_key + ((size_t)b * K_SEQ + k) * DMODEL;
    float dpk = 0.f, dpp = 0.f, dkk = 0.f;
    for (int i = lane; i < DMODEL; i += 32) {
        float pv = p[i], kv = kk[i];
        dpk += pv * kv; dpp += pv * pv; dkk += kv * kv;
    }
    dpk = red16_sum(dpk); dpk += ds_swz_f<16>(dpk);
    dpp = red16_sum(dpp); dpp += ds_swz_f<16>(dpp);
    dkk = red16_sum(dkk); dkk += ds_swz_f<16>(dkk);
    if (lane == 0) {
        float denom = fmaxf(sqrtf(dpp), 1e-8f) * fmaxf(sqrtf(dkk), 1e-8f);
        sim[(size_t)b * K_SEQ + k] = dpk / denom;
    }
}

// ---------------------------------------------------------------------------
// GEMM: C[M,768] = (A[M,768] @ W[768,768]^T + bias) * scale
// A bf16 row-major, W bf16 row-major [out][in] (contiguous inner-k per output
// column, which is exactly the WMMA B-fragment layout).
// Block = 256 thr (8 waves), tile 128(M) x 64(N), k-tile 64.
// Double-buffered pipeline using GLOBAL_LOAD_ASYNC_TO_LDS_B128: the async DMA
// for the next k-tile runs while wmma consumes the current LDS buffer.
// ---------------------------------------------------------------------------
template <bool OUT_BF16>
__global__ __launch_bounds__(256)
void gemm_bias_kernel(const bf16_t* __restrict__ A, const bf16_t* __restrict__ W,
                      const float* __restrict__ bias, void* __restrict__ Cout,
                      int M, float scale) {
    __shared__ bf16_t As[2][128 * 64];   // [m][k]
    __shared__ bf16_t Bs[2][64 * 64];    // [n][k]
    const int mbase = blockIdx.x * 128;
    const int nbase = blockIdx.y * 64;
    const int t = threadIdx.x, wave = t >> 5, lane = t & 31;
    const int lm = lane & 15;
    const bool hi = lane >= 16;
    const int ka = hi ? 16 : 0, kb2 = hi ? 24 : 8, off = hi ? 16 : 0;

    // staging assignments: A 128x64 halfs -> 32 halfs/thread, B 64x64 -> 16
    const int arow = t >> 1, ac = (t & 1) * 32;
    const int brow = t >> 2, bc = (t & 3) * 16;

    auto issue_tile = [&](int kd, int buf) {
        const bf16_t* ag = A + (size_t)(mbase + arow) * DMODEL + kd + ac;
        unsigned al = lds_off32(As[buf] + arow * 64 + ac);
        async_copy_b128(al,      ag);
        async_copy_b128(al + 16, ag + 8);
        async_copy_b128(al + 32, ag + 16);
        async_copy_b128(al + 48, ag + 24);
        const bf16_t* bg = W + (size_t)(nbase + brow) * DMODEL + kd + bc;
        unsigned bl = lds_off32(Bs[buf] + brow * 64 + bc);
        async_copy_b128(bl,      bg);
        async_copy_b128(bl + 16, bg + 8);
    };

    v8f acc[4];
    #pragma unroll
    for (int j = 0; j < 4; ++j)
        acc[j] = (v8f){0.f,0.f,0.f,0.f,0.f,0.f,0.f,0.f};

    const int NSTAGE = DMODEL / 64;   // 12
    issue_tile(0, 0);
    wait_async0();
    __syncthreads();

    for (int s = 0; s < NSTAGE; ++s) {
        const int buf = s & 1;
        if (s + 1 < NSTAGE) issue_tile((s + 1) * 64, (s + 1) & 1);  // DMA in flight
        #pragma unroll
        for (int ks = 0; ks < 2; ++ks) {
            FragB a;
            const bf16_t* ar = As[buf] + (wave * 16 + lm) * 64 + ks * 32;
            a.q[0] = *(const u32x4*)(ar + ka);
            a.q[1] = *(const u32x4*)(ar + kb2);
            #pragma unroll
            for (int j = 0; j < 4; ++j) {
                FragB bf;
                const bf16_t* br = Bs[buf] + (j * 16 + lm) * 64 + ks * 32 + off;
                bf.q[0] = *(const u32x4*)(br);
                bf.q[1] = *(const u32x4*)(br + 8);
                acc[j] = wmma_bf16(a.v, bf.v, acc[j]);
            }
        }
        wait_async0();
        __syncthreads();
    }

    #pragma unroll
    for (int j = 0; j < 4; ++j) {
        int nn = nbase + j * 16 + lm;
        float bsv = bias[nn];
        #pragma unroll
        for (int r = 0; r < 8; ++r) {
            int mm = mbase + wave * 16 + r + (hi ? 8 : 0);
            float v = (acc[j][r] + bsv) * scale;
            if (OUT_BF16)
                ((bf16_t*)Cout)[(size_t)mm * DMODEL + nn] = f2bf(v);
            else
                ((float*)Cout)[(size_t)mm * DMODEL + nn] = v;
        }
    }
}

// ---------------------------------------------------------------------------
// Flash attention with per-head sim bias.
// Grid: (T/64, H, bs), block = 128 (4 waves; each wave owns 16 T rows).
// 64-wide k chunks; K chunk staged via async DMA to LDS, V chunk staged
// through registers with a paired-row transpose (packed ds_store_b32).
// ---------------------------------------------------------------------------
__global__ __launch_bounds__(128)
void attn_kernel(const bf16_t* __restrict__ Q, const bf16_t* __restrict__ Kp,
                 const bf16_t* __restrict__ Vp, const float* __restrict__ sim,
                 const float* __restrict__ beta, bf16_t* __restrict__ Out) {
    __shared__ float  simb[K_SEQ];
    __shared__ bf16_t Kch[2][64 * HDIM];     // [kk][d]
    __shared__ bf16_t Vch[2][HDIM * 64];     // [d][kk]  (transposed)
    __shared__ bf16_t Pbuf[4][16 * 64];      // per-wave P tile [m][kk]

    const int b = blockIdx.z, h = blockIdx.y;
    const int t = threadIdx.x;
    const int wave = t >> 5, lane = t & 31;
    const int m0 = blockIdx.x * 64 + wave * 16;
    const int lm = lane & 15;
    const bool hi = lane >= 16;
    const int ka = hi ? 16 : 0, kb2 = hi ? 24 : 8, off = hi ? 16 : 0;

    float bh = beta[h];
    for (int i = t; i < K_SEQ; i += 128)
        simb[i] = bh * sim[(size_t)b * K_SEQ + i];

    // Q A-fragments for two 32-deep inner-d steps, held for the whole loop
    FragB qa[2];
    {
        const bf16_t* qrow = Q + (size_t)(b * T_SEQ + m0 + lm) * DMODEL + h * HDIM;
        #pragma unroll
        for (int s = 0; s < 2; ++s) {
            qa[s].q[0] = *(const u32x4*)(qrow + s * 32 + ka);
            qa[s].q[1] = *(const u32x4*)(qrow + s * 32 + kb2);
        }
    }

    // K staging: 64 rows x 64 halfs -> 32 halfs (2 lanes per row)
    const int srow = t >> 1, sc = (t & 1) * 32;
    // V staging (transpose): each thread owns 2 adjacent k-rows x 16 d-cols
    const int vk0 = (t & 31) * 2, vc0 = (t >> 5) * 16;
    Stage16 vlo, vhi;

    auto issue_K = [&](int kb, int buf) {
        const bf16_t* g = Kp + (size_t)(b * K_SEQ + kb + srow) * DMODEL + h * HDIM + sc;
        unsigned l = lds_off32(Kch[buf] + srow * HDIM + sc);
        async_copy_b128(l,      g);
        async_copy_b128(l + 16, g + 8);
        async_copy_b128(l + 32, g + 16);
        async_copy_b128(l + 48, g + 24);
    };
    auto load_V = [&](int kb) {
        const bf16_t* v0 = Vp + (size_t)(b * K_SEQ + kb + vk0) * DMODEL + h * HDIM + vc0;
        const bf16_t* v1 = v0 + DMODEL;
        vlo.q[0] = ((const u32x4*)v0)[0]; vlo.q[1] = ((const u32x4*)v0)[1];
        vhi.q[0] = ((const u32x4*)v1)[0]; vhi.q[1] = ((const u32x4*)v1)[1];
    };
    auto store_V = [&](int buf) {
        unsigned int* vd = (unsigned int*)Vch[buf];
        #pragma unroll
        for (int j = 0; j < 16; ++j)   // packed pair {k,k+1} per dword
            vd[(vc0 + j) * 32 + (vk0 >> 1)] =
                (unsigned)vlo.u[j] | ((unsigned)vhi.u[j] << 16);
    };

    float mrow[8], lrow[8];
    v8f o[4];
    #pragma unroll
    for (int r = 0; r < 8; ++r) { mrow[r] = -1e30f; lrow[r] = 0.f; }
    #pragma unroll
    for (int j = 0; j < 4; ++j)
        o[j] = (v8f){0.f,0.f,0.f,0.f,0.f,0.f,0.f,0.f};

    const int NCH = K_SEQ / 64;   // 8
    issue_K(0, 0);
    load_V(0);
    store_V(0);
    wait_async0();
    __syncthreads();

    for (int ch = 0; ch < NCH; ++ch) {
        const int buf = ch & 1;
        const int kb = ch * 64;
        if (ch + 1 < NCH) {                       // overlap with wmma below
            issue_K((ch + 1) * 64, (ch + 1) & 1);
            load_V((ch + 1) * 64);
        }

        // ---- scores: four 16-wide n-tiles over this 64-wide k chunk ----
        v8f scr[4];
        #pragma unroll
        for (int nt = 0; nt < 4; ++nt)
            scr[nt] = (v8f){0.f,0.f,0.f,0.f,0.f,0.f,0.f,0.f};
        #pragma unroll
        for (int s = 0; s < 2; ++s) {
            #pragma unroll
            for (int nt = 0; nt < 4; ++nt) {
                FragB bk;
                const bf16_t* kr = Kch[buf] + (nt * 16 + lm) * HDIM + s * 32 + off;
                bk.q[0] = *(const u32x4*)(kr);
                bk.q[1] = *(const u32x4*)(kr + 8);
                scr[nt] = wmma_bf16(qa[s].v, bk.v, scr[nt]);
            }
        }
        float bias0 = simb[kb + lm];
        float bias1 = simb[kb + 16 + lm];
        float bias2 = simb[kb + 32 + lm];
        float bias3 = simb[kb + 48 + lm];

        // ---- online softmax; rows live in 16-lane half-groups ----
        #pragma unroll
        for (int r = 0; r < 8; ++r) {
            float x0 = scr[0][r] + bias0;
            float x1 = scr[1][r] + bias1;
            float x2 = scr[2][r] + bias2;
            float x3 = scr[3][r] + bias3;
            float tm = fmaxf(fmaxf(x0, x1), fmaxf(x2, x3));
            tm = red16_max(tm);
            float mnew = fmaxf(mrow[r], tm);
            float corr = __expf(mrow[r] - mnew);
            mrow[r] = mnew;
            float p0 = __expf(x0 - mnew);
            float p1 = __expf(x1 - mnew);
            float p2 = __expf(x2 - mnew);
            float p3 = __expf(x3 - mnew);
            float rs = red16_sum((p0 + p1) + (p2 + p3));
            lrow[r] = lrow[r] * corr + rs;
            o[0][r] *= corr; o[1][r] *= corr; o[2][r] *= corr; o[3][r] *= corr;
            int mm = r + (hi ? 8 : 0);
            bf16_t* prow = &Pbuf[wave][mm * 64 + lm];
            prow[0]  = f2bf(p0);
            prow[16] = f2bf(p1);
            prow[32] = f2bf(p2);
            prow[48] = f2bf(p3);
        }
        asm volatile("s_wait_dscnt 0" ::: "memory");   // wave-local P relayout

        // ---- P @ V over this chunk: two 32-deep steps x four n-tiles ----
        #pragma unroll
        for (int s = 0; s < 2; ++s) {
            FragB pa;
            const bf16_t* pr = &Pbuf[wave][lm * 64 + s * 32];
            pa.q[0] = *(const u32x4*)(pr + ka);
            pa.q[1] = *(const u32x4*)(pr + kb2);
            #pragma unroll
            for (int j = 0; j < 4; ++j) {
                FragB vb;
                const bf16_t* vr = Vch[buf] + (j * 16 + lm) * 64 + s * 32 + off;
                vb.q[0] = *(const u32x4*)(vr);
                vb.q[1] = *(const u32x4*)(vr + 8);
                o[j] = wmma_bf16(pa.v, vb.v, o[j]);
            }
        }

        if (ch + 1 < NCH) store_V((ch + 1) & 1);
        wait_async0();
        __syncthreads();
    }

    // ---- normalize and emit bf16 attention output [bs*T, 768] ----
    #pragma unroll
    for (int r = 0; r < 8; ++r) {
        float inv = 1.0f / lrow[r];
        int mm = r + (hi ? 8 : 0);
        bf16_t* orow = Out + (size_t)(b * T_SEQ + m0 + mm) * DMODEL + h * HDIM;
        #pragma unroll
        for (int j = 0; j < 4; ++j)
            orow[j * 16 + lm] = f2bf(o[j][r] * inv);
    }
}

// ---------------------------------------------------------------------------
// Host-side launch
// ---------------------------------------------------------------------------
extern "C" void kernel_launch(void* const* d_in, const int* in_sizes, int n_in,
                              void* d_out, int out_size, void* d_ws, size_t ws_size,
                              hipStream_t stream) {
    const float* hidden   = (const float*)d_in[0];
    /* d_in[1] attention_mask: unused by reference */
    const float* pooled   = (const float*)d_in[2];
    const float* kg_key   = (const float*)d_in[3];
    const float* kg_value = (const float*)d_in[4];
    const float* beta     = (const float*)d_in[5];
    const float* Wq = (const float*)d_in[6];
    const float* bq = (const float*)d_in[7];
    const float* Wk = (const float*)d_in[8];
    const float* bk = (const float*)d_in[9];
    const float* Wv = (const float*)d_in[10];
    const float* bv = (const float*)d_in[11];
    const float* Wo = (const float*)d_in[12];
    const float* bo = (const float*)d_in[13];

    const int MT = 8 * T_SEQ;   // 16384 token rows
    const int MK = 8 * K_SEQ;   // 4096 kg rows

    char* ws = (char*)d_ws;
    size_t woff = 0;
    auto alloc = [&](size_t bytes) {
        char* p = ws + woff;
        woff += (bytes + 255) & ~(size_t)255;
        return p;
    };
    bf16_t* Xh   = (bf16_t*)alloc((size_t)MT * DMODEL * 2);
    bf16_t* Xkg  = (bf16_t*)alloc((size_t)MK * DMODEL * 2);
    bf16_t* Wqb  = (bf16_t*)alloc((size_t)DMODEL * DMODEL * 2);
    bf16_t* Wkb  = (bf16_t*)alloc((size_t)DMODEL * DMODEL * 2);
    bf16_t* Wvb  = (bf16_t*)alloc((size_t)DMODEL * DMODEL * 2);
    bf16_t* Wob  = (bf16_t*)alloc((size_t)DMODEL * DMODEL * 2);
    bf16_t* Qb   = (bf16_t*)alloc((size_t)MT * DMODEL * 2);
    bf16_t* Kb   = (bf16_t*)alloc((size_t)MK * DMODEL * 2);
    bf16_t* Vb   = (bf16_t*)alloc((size_t)MK * DMODEL * 2);
    bf16_t* Attn = (bf16_t*)alloc((size_t)MT * DMODEL * 2);
    float*  Sim  = (float*)alloc((size_t)8 * K_SEQ * 4);
    (void)ws_size; (void)in_sizes; (void)n_in; (void)out_size;

    // 1) converts (all element counts divisible by 1024)
    {
        int n;
        n = MT * DMODEL;     cvt_f32_bf16_kernel<<<n / 1024, 256, 0, stream>>>(hidden,   Xh,  n / 4);
        n = MK * DMODEL;     cvt_f32_bf16_kernel<<<n / 1024, 256, 0, stream>>>(kg_value, Xkg, n / 4);
        n = DMODEL * DMODEL; cvt_f32_bf16_kernel<<<n / 1024, 256, 0, stream>>>(Wq, Wqb, n / 4);
        cvt_f32_bf16_kernel<<<n / 1024, 256, 0, stream>>>(Wk, Wkb, n / 4);
        cvt_f32_bf16_kernel<<<n / 1024, 256, 0, stream>>>(Wv, Wvb, n / 4);
        cvt_f32_bf16_kernel<<<n / 1024, 256, 0, stream>>>(Wo, Wob, n / 4);
    }

    // 2) cosine similarity
    sim_kernel<<<(8 * K_SEQ) / 8, 256, 0, stream>>>(pooled, kg_key, Sim);

    // 3) projections (bf16 out)
    const float scale = 0.125f;   // HD^-0.5
    gemm_bias_kernel<true><<<dim3(MT / 128, DMODEL / 64), 256, 0, stream>>>(Xh,  Wqb, bq, Qb, MT, scale);
    gemm_bias_kernel<true><<<dim3(MK / 128, DMODEL / 64), 256, 0, stream>>>(Xkg, Wkb, bk, Kb, MK, 1.0f);
    gemm_bias_kernel<true><<<dim3(MK / 128, DMODEL / 64), 256, 0, stream>>>(Xkg, Wvb, bv, Vb, MK, 1.0f);

    // 4) attention
    attn_kernel<<<dim3(T_SEQ / 64, NHEAD, 8), 128, 0, stream>>>(Qb, Kb, Vb, Sim, beta, Attn);

    // 5) output projection (f32 out)
    gemm_bias_kernel<false><<<dim3(MT / 128, DMODEL / 64), 256, 0, stream>>>(Attn, Wob, bo, (float*)d_out, MT, 1.0f);
}
